// QuadConvLayer_17660905521726
// MI455X (gfx1250) — matrix operand order repaired and checked by reading, not compiled, not run on hardware
//
#include <hip/hip_runtime.h>

// ---------------------------------------------------------------------------
// QuadConv layer for MI455X (gfx1250), wave32 + v_wmma_f32_16x16x32_bf16.
//
// Shapes: B=16, C_IN=C_OUT=32, N_IN=4096, N_OUT=1024, hidden=64, E=runtime.
// Each workgroup owns 32 eval points: computes its 32x1024 filter tile with
// WMMA into LDS (bf16, never touches HBM), then runs the per-e
// (batch x cin)@(cin x cout) GEMMs with WMMA and scrambled atomic scatter.
// 32-e tiling halves the L2 re-read of W3 vs a 16-e tile (each B fragment
// feeds two A tiles).
// ---------------------------------------------------------------------------

typedef __attribute__((ext_vector_type(16))) __bf16 bf16x16;
typedef __attribute__((ext_vector_type(8)))  float  f32x8;

union ABFrag {
    bf16x16        v;      // WMMA A/B operand: 8 VGPRs, 16 bf16 per lane
    uint4          q[2];
    unsigned int   u[8];
    unsigned short s[16];
};

// float -> bf16 bits, round-to-nearest-even (pure bit math)
static __device__ __forceinline__ unsigned short bf16_bits(float f) {
    union { float f; unsigned int u; } c; c.f = f;
    unsigned int u = c.u;
    u += 0x7fffu + ((u >> 16) & 1u);
    return (unsigned short)(u >> 16);
}

// ---------------------------------------------------------------------------
// P1: H[e][t] = bf16( sin( sum_j sin(locs[e]@W1)[j] * W2[j][t] ) ), t in 0..63
// ---------------------------------------------------------------------------
__global__ void k_filter_mlp(const float* __restrict__ locs,
                             const float* __restrict__ W1,
                             const float* __restrict__ W2,
                             unsigned short* __restrict__ Hh,
                             int E) {
    __shared__ float h1[64];
    const int e = blockIdx.x;
    const int t = threadIdx.x;            // 0..63
    const float lx = locs[2 * e + 0];
    const float ly = locs[2 * e + 1];
    h1[t] = __builtin_sinf(lx * W1[t] + ly * W1[64 + t]);
    __syncthreads();
    float acc = 0.0f;
#pragma unroll 16
    for (int j = 0; j < 64; ++j) acc += h1[j] * W2[j * 64 + t];
    Hh[e * 64 + t] = bf16_bits(__builtin_sinf(acc));
}

// ---------------------------------------------------------------------------
// P2: repack W3 (64x1024 f32, row-major) into WMMA-B lane layout, bf16 pairs.
// For n-tile t (16 cols), K-step kk (K base 32*kk), lane L, vgpr v:
//   dword = pack( W3[K][n], W3[K+1][n] ),  K = 32*kk + 2*v + 16*(L>>4),
//   n = 16*t + (L&15).  Flat dword index == thread id by construction.
// ---------------------------------------------------------------------------
__global__ void k_pack_w3(const float* __restrict__ W3,
                          unsigned int* __restrict__ W3p) {
    const int id = blockIdx.x * blockDim.x + threadIdx.x;  // 0..32767
    const int v  = id & 7;
    const int L  = (id >> 3) & 31;
    const int kk = (id >> 8) & 1;
    const int t  = id >> 9;
    const int K  = 32 * kk + 2 * v + 16 * (L >> 4);
    const int n  = 16 * t + (L & 15);
    const unsigned int lo = bf16_bits(W3[K * 1024 + n]);
    const unsigned int hi = bf16_bits(W3[(K + 1) * 1024 + n]);
    W3p[id] = (hi << 16) | lo;
}

// ---------------------------------------------------------------------------
// P3: precompute the reference's raw-reshape index scramble.
// f in [0, 32E):  c = f/E, s = f%E
//   gather index  gidx[f] = c*4096 + src[s]   (into features[b][.][.])
//   scatter index sidx[f] = c*1024 + seg[s]   (into out[b][.][.])
// ---------------------------------------------------------------------------
__global__ void k_indices(const long long* __restrict__ ei,  // [E][2] = (seg, src)
                          int* __restrict__ gidx,
                          int* __restrict__ sidx,
                          int E) {
    const int f = blockIdx.x * blockDim.x + threadIdx.x;
    if (f >= 32 * E) return;
    const int c = f / E;
    const int s = f - c * E;
    const int seg = (int)ei[2 * s + 0];
    const int src = (int)ei[2 * s + 1];
    gidx[f] = c * 4096 + src;
    sidx[f] = c * 1024 + seg;
}

// ---------------------------------------------------------------------------
// Main fused kernel: one workgroup (256 thr = 8 wave32) per 32 eval points.
//  Phase 2: Filt(32x1024) = H_tile(32x64) @ W3  via wmma bf16, -> LDS [e][j][i]
//  Phase 3: per e: vals(16b x 32j) = g(16b x 32i) @ filt_e(32i x 32j) via wmma,
//           then scrambled atomic scatter-add into out.
// ---------------------------------------------------------------------------
__global__ void __launch_bounds__(256)
k_quadconv(const float* __restrict__ feats,          // [16][32][4096]
           const unsigned short* __restrict__ Hh,    // [rows32][64] bf16 bits
           const uint4* __restrict__ W3p,            // packed B operands
           const int* __restrict__ gidx,
           const int* __restrict__ sidx,
           float* __restrict__ out,                  // [16][32][1024]
           int E) {
    __shared__ unsigned short sfilt[32 * 1024];      // [e_local][j=cout][i=cin], 64 KB

    const int tid  = threadIdx.x;
    const int lane = tid & 31;
    const int wave = tid >> 5;          // 0..7
    const int hi   = lane >> 4;         // 0/1
    const int m    = lane & 15;
    const int e0   = blockIdx.x * 32;

    // Prefetch this block's gather/scatter index slices (overlaps phase 2);
    // emits global_prefetch_b8.
    {
        const int pf = e0 * 32 + tid * 4;            // 1024 ints per block
        if (pf < 32 * E) {
            __builtin_prefetch(gidx + pf, 0, 0);
            __builtin_prefetch(sidx + pf, 0, 0);
        }
    }

    // ---------- Phase 2: filter-tile GEMM --------------------------------
    // A operand (contiguous k order, same as B): lane row = e0 + 16*et + m,
    //   vgpr v -> K = 2v + 16*hi (+32*kk).
    ABFrag A[2][2];                                   // [e-tile][kk]
#pragma unroll
    for (int et = 0; et < 2; ++et) {
        const unsigned short* hrow = Hh + (size_t)(e0 + 16 * et + m) * 64;
#pragma unroll
        for (int kk = 0; kk < 2; ++kk) {
            A[et][kk].q[0] = *(const uint4*)(hrow + kk * 32 + hi * 16);       // v0..v3
            A[et][kk].q[1] = *(const uint4*)(hrow + kk * 32 + hi * 16 + 8);  // v4..v7
        }
    }
#pragma unroll
    for (int qt = 0; qt < 8; ++qt) {
        const int t = wave * 8 + qt;                 // n-tile 0..63
        f32x8 acc[2] = {{}, {}};
#pragma unroll
        for (int kk = 0; kk < 2; ++kk) {
            ABFrag B;                                 // loaded once, feeds 2 A tiles
            const uint4* bp = W3p + (size_t)(((t * 2 + kk) * 32 + lane) * 2);
            B.q[0] = bp[0];
            B.q[1] = bp[1];
#pragma unroll
            for (int et = 0; et < 2; ++et)
                acc[et] = __builtin_amdgcn_wmma_f32_16x16x32_bf16(
                    false, A[et][kk].v, false, B.v, (short)0, acc[et], false, false);
        }
        // C layout: vgpr v -> e_row = 16*et + v + 8*hi ; col n_f = 16*t + m.
        // filt[e][i][j] with i = n_f>>5 (cin), j = n_f&31 (cout); store [e][j][i].
        const int nf   = t * 16 + m;
        const int base = (nf & 31) * 32 + (nf >> 5);
#pragma unroll
        for (int et = 0; et < 2; ++et)
#pragma unroll
            for (int v = 0; v < 8; ++v)
                sfilt[(16 * et + v + 8 * hi) * 1024 + base] = bf16_bits(acc[et][v]);
    }
    __syncthreads();

    // ---------- Phase 3: per-e batch GEMM + scatter ----------------------
#pragma unroll
    for (int sub = 0; sub < 4; ++sub) {
        const int el = wave * 4 + sub;               // e_local 0..31 (wave-uniform)
        const int e  = e0 + el;
        if (e < E) {                                  // wave-uniform: EXEC all-1s inside
            // A operand: g[b][i], rows = batch = m, K = cin (scrambled gather)
            ABFrag G;
            const float* fb = feats + (size_t)m * (32 * 4096);
            const int*   gi = gidx + (size_t)e * 32;
#pragma unroll
            for (int v = 0; v < 8; ++v) {
                const int i = 2 * v + 16 * hi;
                G.s[2 * v + 0] = bf16_bits(fb[gi[i]]);
                G.s[2 * v + 1] = bf16_bits(fb[gi[i + 1]]);
            }
            const int* si = sidx + (size_t)e * 32;
#pragma unroll
            for (int nt = 0; nt < 2; ++nt) {
                // B operand from LDS: n = 16*nt + m, vgpr v -> K pair (2v+16hi, +1)
                ABFrag B;
                const unsigned short* lp =
                    sfilt + el * 1024 + (nt * 16 + m) * 32 + hi * 16;
                B.q[0] = *(const uint4*)(lp);
                B.q[1] = *(const uint4*)(lp + 8);
                f32x8 acc = {};
                acc = __builtin_amdgcn_wmma_f32_16x16x32_bf16(
                    false, G.v, false, B.v, (short)0, acc, false, false);
                // C: vgpr v -> batch b = v + 8*hi ; col j = 16*nt + m.
                const int j  = nt * 16 + m;
                const int oi = si[j];                 // c_out*1024 + seg (scrambled)
                float* ob = out + oi;
#pragma unroll
                for (int v = 0; v < 8; ++v)
                    atomicAdd(ob + (size_t)(v + 8 * hi) * (32 * 1024), acc[v]);
            }
        }
    }
}

// ---------------------------------------------------------------------------
extern "C" void kernel_launch(void* const* d_in, const int* in_sizes, int n_in,
                              void* d_out, int out_size, void* d_ws, size_t ws_size,
                              hipStream_t stream) {
    const float*     feats = (const float*)d_in[0];  // (16,32,4096)
    const float*     W1    = (const float*)d_in[1];  // (2,64)
    const float*     W2    = (const float*)d_in[2];  // (64,64)
    const float*     W3    = (const float*)d_in[3];  // (64,1024)
    const float*     locs  = (const float*)d_in[4];  // (E,2)
    const long long* ei    = (const long long*)d_in[5];  // (E,2) int64

    const int E      = in_sizes[4] / 2;
    const int rows32 = ((E + 31) / 32) * 32;

    // Workspace carve-up (256B aligned regions)
    char*  ws  = (char*)d_ws;
    size_t off = 0;
    unsigned short* Hh = (unsigned short*)(ws + off);
    off += (size_t)rows32 * 64 * 2;          off = (off + 255) & ~(size_t)255;
    unsigned int* W3p = (unsigned int*)(ws + off);
    off += (size_t)32768 * 4;                off = (off + 255) & ~(size_t)255;
    int* gidx = (int*)(ws + off);
    off += (size_t)32 * E * 4;               off = (off + 255) & ~(size_t)255;
    int* sidx = (int*)(ws + off);
    (void)ws_size; (void)n_in;

    hipMemsetAsync(d_out, 0, (size_t)out_size * sizeof(float), stream);

    k_filter_mlp<<<E, 64, 0, stream>>>(locs, W1, W2, Hh, E);
    k_pack_w3<<<128, 256, 0, stream>>>(W3, W3p);
    k_indices<<<(32 * E + 255) / 256, 256, 0, stream>>>(ei, gidx, sidx, E);

    const int nTiles = (E + 31) / 32;
    k_quadconv<<<nTiles, 256, 0, stream>>>(feats, Hh, (const uint4*)W3p,
                                           gidx, sidx, (float*)d_out, E);
}
